// LSTMModel_67680094650911
// MI455X (gfx1250) — matrix operand order repaired and checked by reading, not compiled, not run on hardware
//
#include <hip/hip_runtime.h>
#include <hip/hip_bf16.h>

// ---------------------------------------------------------------------------
// Fused 2-layer LSTM + FC head for gfx1250 (CDNA5, wave32, WMMA).
//
// One block = 16 batch rows (one WMMA M tile), 256 threads = 8 waves.
// Weights are pre-formatted into per-wave WMMA B fragments held in VGPRs for
// the whole T=168 scan. h/c state lives in LDS; matmul operands are f16 with
// f32 WMMA accumulation; cell state & activations are f32. Biases are folded
// into the WMMA accumulator init. Activations use branch-free hardware
// exp2/rcp. x loads are software-pipelined one timestep ahead.
// ---------------------------------------------------------------------------

typedef _Float16 v16h __attribute__((ext_vector_type(16)));
typedef _Float16 v8h  __attribute__((ext_vector_type(8)));
typedef _Float16 v4h  __attribute__((ext_vector_type(4)));
typedef _Float16 v2h  __attribute__((ext_vector_type(2)));
typedef float    v8f  __attribute__((ext_vector_type(8)));
typedef float    v4f  __attribute__((ext_vector_type(4)));
typedef float    v2f  __attribute__((ext_vector_type(2)));

#define B_TOT 4096
#define T_LEN 168
#define IN_D  16
#define H1D   64
#define H2D   32
#define G1    256  // 4*H1
#define G2    128  // 4*H2
#define KC1   96   // 16 (x) + 64 (h1) + 16 (pad), 3 chunks of 32
#define KC2   96   // 64 (h1) + 32 (h2),          3 chunks of 32

union AF { v16h v; v8h h[2]; };

#define LOG2E  1.4426950408889634f
#define LOG2E2 2.8853900817779268f

// Branch-free activations built on v_exp_f32 (exp2) + v_rcp_f32.
// Saturate correctly for large |x| (exp2 -> inf => rcp -> 0; exp2 -> 0 => rcp(1)).
__device__ __forceinline__ float fast_sigmoid(float x) {
  return __builtin_amdgcn_rcpf(1.0f + __builtin_amdgcn_exp2f(-LOG2E * x));
}
__device__ __forceinline__ float fast_tanh(float x) {
  return 1.0f - 2.0f * __builtin_amdgcn_rcpf(1.0f + __builtin_amdgcn_exp2f(LOG2E2 * x));
}

// B-fragment (32x16 K-major chunk) for layer1 combined weights
// [W_ih1 (K 0..15) | W_hh1 (K 16..79) | zero (K 80..95)], row g, K run kbase..kbase+15.
__device__ __forceinline__ v16h load_w_frag_l1(const float* __restrict__ Wih,
                                               const float* __restrict__ Whh,
                                               int g, int kbase) {
  v16h r;
  if (kbase >= 80) {
#pragma unroll
    for (int i = 0; i < 16; ++i) r[i] = (_Float16)0.0f;
  } else if (kbase == 0) {
    const float* p = Wih + g * IN_D;
#pragma unroll
    for (int i = 0; i < 16; ++i) r[i] = (_Float16)p[i];
  } else {
    const float* p = Whh + g * H1D + (kbase - 16);
#pragma unroll
    for (int i = 0; i < 16; ++i) r[i] = (_Float16)p[i];
  }
  return r;
}

// Layer2 combined weights: [W_ih2 (K 0..63) | W_hh2 (K 64..95)].
__device__ __forceinline__ v16h load_w_frag_l2(const float* __restrict__ Wih,
                                               const float* __restrict__ Whh,
                                               int g, int kbase) {
  v16h r;
  if (kbase < 64) {
    const float* p = Wih + g * H1D + kbase;
#pragma unroll
    for (int i = 0; i < 16; ++i) r[i] = (_Float16)p[i];
  } else {
    const float* p = Whh + g * H2D + (kbase - 64);
#pragma unroll
    for (int i = 0; i < 16; ++i) r[i] = (_Float16)p[i];
  }
  return r;
}

__global__ __launch_bounds__(256)
void lstm2_fused_kernel(const float* __restrict__ x,
                        const float* __restrict__ W_ih1, const float* __restrict__ W_hh1,
                        const float* __restrict__ b_ih1, const float* __restrict__ b_hh1,
                        const float* __restrict__ W_ih2, const float* __restrict__ W_hh2,
                        const float* __restrict__ b_ih2, const float* __restrict__ b_hh2,
                        const float* __restrict__ W_fc,  const float* __restrict__ b_fc,
                        float* __restrict__ out)
{
  __shared__ _Float16 Abuf1[16 * KC1];   // [x_t | h1 | pad] f16, A staging L1
  __shared__ _Float16 Abuf2[16 * KC2];   // [h1 | h2]        f16, A staging L2
  __shared__ float    Gbuf[16 * G1];     // gate accumulators (f32), reused by L1/L2
  __shared__ float    C1[16 * H1D];      // cell state layer1
  __shared__ float    C2[16 * H2D];      // cell state layer2
  __shared__ float    H2f[16 * H2D];     // h2 in f32 (for FC head)

  const int tid    = threadIdx.x;
  const int lane   = tid & 31;
  const int wave   = tid >> 5;            // 0..7
  const int ncol   = lane & 15;           // B/D column within tile
  const int mrow   = lane & 15;           // A row this lane holds
  const int klo    = (lane < 16) ? 0 : 16; // A-fragment K sub-offset
  const int kbB    = (lane < 16) ? 0 : 16;// B-fragment K base offset
  const int base_b = blockIdx.x * 16;

  const int ta  = (2 * wave) * 16;        // L1 gate column base, tile A
  const int tb  = (2 * wave + 1) * 16;    // L1 gate column base, tile B
  const int tb2 = wave * 16;              // L2 gate column base

  // ---------------- init LDS state ----------------
  for (int i = tid; i < 16 * KC1; i += 256) Abuf1[i] = (_Float16)0.0f;
  for (int i = tid; i < 16 * KC2; i += 256) Abuf2[i] = (_Float16)0.0f;
  for (int i = tid; i < 16 * H1D; i += 256) C1[i] = 0.0f;
  for (int i = tid; i < 16 * H2D; i += 256) { C2[i] = 0.0f; H2f[i] = 0.0f; }

  // ---------------- per-lane bias (folded into WMMA C init) ----------------
  const float bias_a  = b_ih1[ta  + ncol] + b_hh1[ta  + ncol];
  const float bias_b  = b_ih1[tb  + ncol] + b_hh1[tb  + ncol];
  const float bias_2  = b_ih2[tb2 + ncol] + b_hh2[tb2 + ncol];

  // ---------------- persistent weight fragments (VGPR-resident) ----------------
  v16h B1a[3], B1b[3], B2w[3];
  {
    const int g1a = ta  + ncol;
    const int g1b = tb  + ncol;
    const int g2  = tb2 + ncol;
#pragma unroll
    for (int c = 0; c < 3; ++c) {
      const int kb = 32 * c + kbB;
      B1a[c] = load_w_frag_l1(W_ih1, W_hh1, g1a, kb);
      B1b[c] = load_w_frag_l1(W_ih1, W_hh1, g1b, kb);
      B2w[c] = load_w_frag_l2(W_ih2, W_hh2, g2,  kb);
    }
  }
  __syncthreads();

  const int xm = tid >> 4;                 // 0..15: batch row within tile
  const int xd = tid & 15;                 // 0..15: input dim
  const float* xrow = x + (size_t)(base_b + xm) * T_LEN * IN_D + xd;

  const int em = tid >> 4;                 // elementwise: batch row
  const int j4 = (tid & 15) * 4;           // elementwise L1: 4 hidden units
  const int j2 = (tid & 15) * 2;           // elementwise L2: 2 hidden units

  // ---- prologue: stage x_0, pipeline x_1 into a register ----
  Abuf1[xm * KC1 + xd] = (_Float16)xrow[0];
  float xv_next = xrow[(T_LEN > 1 ? 1 : 0) * IN_D];
  __syncthreads();

#pragma unroll 1
  for (int t = 0; t < T_LEN; ++t) {
    // ---- phase A: layer1 WMMA: gates1 = [x_t|h1] * W1cat^T + bias1 ----
    {
      AF a0, a1, a2;
      const _Float16* ap = Abuf1 + mrow * KC1;
      a0.h[0] = *(const v8h*)(ap + 0  + klo);
      a0.h[1] = *(const v8h*)(ap + 8  + klo);
      a1.h[0] = *(const v8h*)(ap + 32 + klo);
      a1.h[1] = *(const v8h*)(ap + 40 + klo);
      a2.h[0] = *(const v8h*)(ap + 64 + klo);
      a2.h[1] = *(const v8h*)(ap + 72 + klo);

      v8f acc_a, acc_b;
#pragma unroll
      for (int i = 0; i < 8; ++i) { acc_a[i] = bias_a; acc_b[i] = bias_b; }
      acc_a = __builtin_amdgcn_wmma_f32_16x16x32_f16(false, a0.v, false, B1a[0], (short)0, acc_a, false, false);
      acc_b = __builtin_amdgcn_wmma_f32_16x16x32_f16(false, a0.v, false, B1b[0], (short)0, acc_b, false, false);
      acc_a = __builtin_amdgcn_wmma_f32_16x16x32_f16(false, a1.v, false, B1a[1], (short)0, acc_a, false, false);
      acc_b = __builtin_amdgcn_wmma_f32_16x16x32_f16(false, a1.v, false, B1b[1], (short)0, acc_b, false, false);
      acc_a = __builtin_amdgcn_wmma_f32_16x16x32_f16(false, a2.v, false, B1a[2], (short)0, acc_a, false, false);
      acc_b = __builtin_amdgcn_wmma_f32_16x16x32_f16(false, a2.v, false, B1b[2], (short)0, acc_b, false, false);

#pragma unroll
      for (int i = 0; i < 8; ++i) {
        const int m = (lane < 16) ? i : (i + 8);
        Gbuf[m * G1 + ta + ncol] = acc_a[i];
        Gbuf[m * G1 + tb + ncol] = acc_b[i];
      }
    }
    __syncthreads();

    // ---- phase B: layer1 elementwise + commit pipelined x_{t+1} ----
    {
      // commit x_{t+1} (loaded last iteration / prologue) and start x_{t+2}
      Abuf1[xm * KC1 + xd] = (_Float16)xv_next;
      const int tnn = (t + 2 < T_LEN) ? (t + 2) : (T_LEN - 1);
      xv_next = xrow[(size_t)tnn * IN_D];

      const float* gp = Gbuf + em * G1;
      v4f gi = *(const v4f*)(gp + j4);
      v4f gf = *(const v4f*)(gp + 64 + j4);
      v4f gg = *(const v4f*)(gp + 128 + j4);
      v4f go = *(const v4f*)(gp + 192 + j4);
      v4f cc = *(const v4f*)(C1 + em * H1D + j4);
      v4h hh;
#pragma unroll
      for (int k = 0; k < 4; ++k) {
        const float iv = fast_sigmoid(gi[k]);
        const float fv = fast_sigmoid(gf[k]);
        const float gv = fast_tanh(gg[k]);
        const float ov = fast_sigmoid(go[k]);
        const float cv = fv * cc[k] + iv * gv;
        cc[k] = cv;
        hh[k] = (_Float16)(ov * fast_tanh(cv));
      }
      *(v4f*)(C1 + em * H1D + j4) = cc;
      *(v4h*)(Abuf1 + em * KC1 + 16 + j4) = hh;  // h1 -> L1 A (next step)
      *(v4h*)(Abuf2 + em * KC2 + j4)      = hh;  // h1 -> L2 A (this step)
    }
    __syncthreads();

    // ---- phase C: layer2 WMMA: gates2 = [h1|h2] * W2cat^T + bias2 ----
    {
      AF a0, a1, a2;
      const _Float16* ap = Abuf2 + mrow * KC2;
      a0.h[0] = *(const v8h*)(ap + 0  + klo);
      a0.h[1] = *(const v8h*)(ap + 8  + klo);
      a1.h[0] = *(const v8h*)(ap + 32 + klo);
      a1.h[1] = *(const v8h*)(ap + 40 + klo);
      a2.h[0] = *(const v8h*)(ap + 64 + klo);
      a2.h[1] = *(const v8h*)(ap + 72 + klo);

      v8f acc;
#pragma unroll
      for (int i = 0; i < 8; ++i) acc[i] = bias_2;
      acc = __builtin_amdgcn_wmma_f32_16x16x32_f16(false, a0.v, false, B2w[0], (short)0, acc, false, false);
      acc = __builtin_amdgcn_wmma_f32_16x16x32_f16(false, a1.v, false, B2w[1], (short)0, acc, false, false);
      acc = __builtin_amdgcn_wmma_f32_16x16x32_f16(false, a2.v, false, B2w[2], (short)0, acc, false, false);

#pragma unroll
      for (int i = 0; i < 8; ++i) {
        const int m = (lane < 16) ? i : (i + 8);
        Gbuf[m * G1 + tb2 + ncol] = acc[i];   // reuse Gbuf, stride G1, cols 0..127
      }
    }
    __syncthreads();

    // ---- phase D: layer2 elementwise (-> c2, h2) ----
    {
      const float* gp = Gbuf + em * G1;
      v2f gi = *(const v2f*)(gp + j2);
      v2f gf = *(const v2f*)(gp + 32 + j2);
      v2f gg = *(const v2f*)(gp + 64 + j2);
      v2f go = *(const v2f*)(gp + 96 + j2);
      v2f cc = *(const v2f*)(C2 + em * H2D + j2);
      v2h hh; v2f hf;
#pragma unroll
      for (int k = 0; k < 2; ++k) {
        const float iv = fast_sigmoid(gi[k]);
        const float fv = fast_sigmoid(gf[k]);
        const float gv = fast_tanh(gg[k]);
        const float ov = fast_sigmoid(go[k]);
        const float cv = fv * cc[k] + iv * gv;
        const float hv = ov * fast_tanh(cv);
        cc[k] = cv;
        hf[k] = hv;
        hh[k] = (_Float16)hv;
      }
      *(v2f*)(C2  + em * H2D + j2) = cc;
      *(v2f*)(H2f + em * H2D + j2) = hf;
      *(v2h*)(Abuf2 + em * KC2 + 64 + j2) = hh;  // h2 -> L2 A (next step)
    }
    __syncthreads();
  }

  // ---- FC head: out[b] = h2_last . W_fc + b_fc ----
  if (tid < 16) {
    float acc = b_fc[0];
#pragma unroll
    for (int j = 0; j < H2D; ++j) acc += H2f[tid * H2D + j] * W_fc[j];
    out[base_b + tid] = acc;
  }
}

extern "C" void kernel_launch(void* const* d_in, const int* in_sizes, int n_in,
                              void* d_out, int out_size, void* d_ws, size_t ws_size,
                              hipStream_t stream) {
  (void)in_sizes; (void)n_in; (void)out_size; (void)d_ws; (void)ws_size;
  const float* x    = (const float*)d_in[0];
  const float* Wih1 = (const float*)d_in[1];
  const float* Whh1 = (const float*)d_in[2];
  const float* bih1 = (const float*)d_in[3];
  const float* bhh1 = (const float*)d_in[4];
  const float* Wih2 = (const float*)d_in[5];
  const float* Whh2 = (const float*)d_in[6];
  const float* bih2 = (const float*)d_in[7];
  const float* bhh2 = (const float*)d_in[8];
  const float* Wfc  = (const float*)d_in[9];
  const float* bfc  = (const float*)d_in[10];
  float* out = (float*)d_out;

  dim3 grid(B_TOT / 16);
  dim3 block(256);
  hipLaunchKernelGGL(lstm2_fused_kernel, grid, block, 0, stream,
                     x, Wih1, Whh1, bih1, bhh1,
                     Wih2, Whh2, bih2, bhh2, Wfc, bfc, out);
}
